// AttentionHead_76115410419972
// MI455X (gfx1250) — matrix operand order repaired and checked by reading, not compiled, not run on hardware
//
#include <hip/hip_runtime.h>
#include <hip/hip_bf16.h>

// ---------------------------------------------------------------------------
// Fused causal attention head for MI455X (gfx1250, wave32, WMMA).
//   B=2048, T=128, E=256, H=64, fp32 in/out, f16 WMMA compute (f32 accum).
// One workgroup (256 threads = 8 wave32) per batch element.
// ---------------------------------------------------------------------------

#define B_DIM 2048
#define T_DIM 128
#define E_DIM 256
#define H_DIM 64
#define DROP_KEEP_THRESH 3435973836u   /* floor(0.8 * 2^32) */
#define DROP_SCALE 1.25f               /* 1 / (1 - 0.2) */
#define ATT_SCALE 0.125f               /* H^-0.5 = 1/8 */

typedef _Float16 half8  __attribute__((ext_vector_type(8)));
typedef _Float16 half16 __attribute__((ext_vector_type(16)));
typedef float    floatx8 __attribute__((ext_vector_type(8)));

#define WMMA_F16(a, b, c) \
    __builtin_amdgcn_wmma_f32_16x16x32_f16(false, (a), false, (b), (short)0, (c), false, false)

// ---- A-matrix fragment (16x32 f16) from row-major f16 LDS tile -------------
// lanes 0-15 row M=lane, K = {0..15}; lanes 16-31 row M=lane-16,
// K = {16..31}.  Two contiguous 16B chunks per lane.
__device__ __forceinline__ half16 load_a_frag(const _Float16* base, int row0, int k0,
                                              int stride, int lmod, int lane_half) {
    const _Float16* p = base + (row0 + lmod) * stride + k0 + lane_half * 16;
    half8 lo = *(const half8*)p;
    half8 hi = *(const half8*)(p + 8);
    return __builtin_shufflevector(lo, hi, 0, 1, 2, 3, 4, 5, 6, 7,
                                           8, 9, 10, 11, 12, 13, 14, 15);
}

__device__ __forceinline__ unsigned hash_u32(unsigned x) {
    x ^= x >> 16; x *= 0x7feb352du;
    x ^= x >> 15; x *= 0x846ca68bu;
    x ^= x >> 16;
    return x;
}

// ---------------------------------------------------------------------------
// Kernel 0: convert weights fp32 [E,H] -> f16 transposed WT[3][H][E] in ws.
// Transposed so B-matrix fragments (B[K=e][N=h] = WT[h][e]) are contiguous.
// ---------------------------------------------------------------------------
__global__ void prep_weights_kernel(const float* __restrict__ Wq,
                                    const float* __restrict__ Wk,
                                    const float* __restrict__ Wv,
                                    _Float16* __restrict__ wT) {
    int i = blockIdx.x * blockDim.x + threadIdx.x;   // 0 .. 3*H*E-1
    int j   = i / (H_DIM * E_DIM);
    int rem = i % (H_DIM * E_DIM);
    int h = rem / E_DIM;
    int e = rem % E_DIM;
    const float* W = (j == 0) ? Wq : (j == 1) ? Wk : Wv;
    wT[i] = (_Float16)W[e * H_DIM + h];
}

// ---------------------------------------------------------------------------
// Kernel 1: fused attention, one block per batch element.
// ---------------------------------------------------------------------------
__global__ __launch_bounds__(256) void attention_kernel(const float* __restrict__ x,
                                                        const _Float16* __restrict__ wT,
                                                        float* __restrict__ out) {
    extern __shared__ _Float16 smem_h[];
    _Float16* qs  = smem_h;                 // [128][64]  q row-major
    _Float16* ks  = qs  + T_DIM * H_DIM;    // [128][64]  k row-major
    _Float16* vts = ks  + T_DIM * H_DIM;    // [64][128]  v transposed
    _Float16* ps  = vts + H_DIM * T_DIM;    // [128][128] P (phase 3); xs chunk (phase 1)

    const int b         = blockIdx.x;
    const int tid       = threadIdx.x;
    const int wave      = tid >> 5;         // 0..7 -> rows 16*wave .. 16*wave+15
    const int lane      = tid & 31;
    const int lmod      = lane & 15;
    const int lane_half = lane >> 4;

    // ------------------- Phase 1: q,k,v = x_b @ {Wq,Wk,Wv} -------------------
    floatx8 accQ[4] = {}, accK[4] = {}, accV[4] = {};
    _Float16* xs = ps;                      // reuse P buffer: [128][64] f16 chunk

    for (int ec = 0; ec < 4; ++ec) {        // E chunks of 64
        __syncthreads();
        // stage x[b, :, ec*64 : ec*64+64] fp32 -> f16 LDS (8 elems/thread/iter)
        #pragma unroll
        for (int it = 0; it < 4; ++it) {
            int idx = (tid + it * 256) * 8; // 0..8191 step 8
            int row = idx >> 6;
            int col = idx & 63;
            const float* src = x + ((size_t)b * T_DIM + row) * E_DIM + ec * 64 + col;
            float4 f0 = *(const float4*)src;
            float4 f1 = *(const float4*)(src + 4);
            half8 h;
            h[0] = (_Float16)f0.x; h[1] = (_Float16)f0.y;
            h[2] = (_Float16)f0.z; h[3] = (_Float16)f0.w;
            h[4] = (_Float16)f1.x; h[5] = (_Float16)f1.y;
            h[6] = (_Float16)f1.z; h[7] = (_Float16)f1.w;
            *(half8*)(xs + row * 64 + col) = h;
        }
        if (ec < 3 && tid < 128) {          // prefetch next chunk's rows
            __builtin_prefetch(x + ((size_t)b * T_DIM + tid) * E_DIM + (ec + 1) * 64, 0, 0);
        }
        __syncthreads();

        #pragma unroll
        for (int kk = 0; kk < 2; ++kk) {    // K steps of 32 within chunk
            half16 a = load_a_frag(xs, 16 * wave, kk * 32, 64, lmod, lane_half);
            int eg = ec * 64 + kk * 32;     // global E offset of this K step
            #pragma unroll
            for (int nt = 0; nt < 4; ++nt) {
                // B[K=e][N=h] = WT[j][h][e]; lane holds N = nt*16+lmod,
                // K = eg + lane_half*16 + j  -> contiguous 16 halves (32B).
                const int hcol = nt * 16 + lmod;
                const int eoff = eg + lane_half * 16;
                half16 bq = *(const half16*)(wT + ((0 * H_DIM + hcol) * E_DIM) + eoff);
                half16 bk = *(const half16*)(wT + ((1 * H_DIM + hcol) * E_DIM) + eoff);
                half16 bv = *(const half16*)(wT + ((2 * H_DIM + hcol) * E_DIM) + eoff);
                accQ[nt] = WMMA_F16(a, bq, accQ[nt]);
                accK[nt] = WMMA_F16(a, bk, accK[nt]);
                accV[nt] = WMMA_F16(a, bv, accV[nt]);
            }
        }
    }

    // D layout (16x16 f32): element g <-> (M = g + 8*lane_half, N = lmod)
    #pragma unroll
    for (int nt = 0; nt < 4; ++nt) {
        #pragma unroll
        for (int g = 0; g < 8; ++g) {
            int m = 16 * wave + g + 8 * lane_half;
            int n = nt * 16 + lmod;
            qs[m * H_DIM + n]  = (_Float16)accQ[nt][g];
            ks[m * H_DIM + n]  = (_Float16)accK[nt][g];
            vts[n * T_DIM + m] = (_Float16)accV[nt][g];   // transposed store
        }
    }
    __syncthreads();

    // ------------------- Phase 2: S = q k^T, mask, softmax, dropout ----------
    half16 aq[2];
    #pragma unroll
    for (int kk = 0; kk < 2; ++kk)
        aq[kk] = load_a_frag(qs, 16 * wave, kk * 32, H_DIM, lmod, lane_half);

    floatx8 s[8];
    #pragma unroll
    for (int st = 0; st < 8; ++st) {        // S column tiles (128 cols)
        floatx8 acc = {};
        #pragma unroll
        for (int kk = 0; kk < 2; ++kk) {
            // B[K=h][N=s] = k[s][h]: lane N = st*16+lmod, K = kk*32+lane_half*16+j
            half16 bb = *(const half16*)(ks + (st * 16 + lmod) * H_DIM
                                            + kk * 32 + lane_half * 16);
            acc = WMMA_F16(aq[kk], bb, acc);
        }
        s[st] = acc;
    }

    const float NEG_INF = -__builtin_inff();
    #pragma unroll
    for (int g = 0; g < 8; ++g) {
        const int t = 16 * wave + g + 8 * lane_half;   // global query row
        float rmax = NEG_INF;
        #pragma unroll
        for (int st = 0; st < 8; ++st) {
            int scol = st * 16 + lmod;
            float v = s[st][g] * ATT_SCALE;
            v = (scol <= t) ? v : NEG_INF;             // causal mask
            s[st][g] = v;
            rmax = fmaxf(rmax, v);
        }
        #pragma unroll
        for (int off = 1; off < 16; off <<= 1)         // reduce over 16-lane half
            rmax = fmaxf(rmax, __shfl_xor(rmax, off, 32));

        float rsum = 0.0f;
        #pragma unroll
        for (int st = 0; st < 8; ++st) {
            float e = __expf(s[st][g] - rmax);         // exp(-inf)=0 for masked
            s[st][g] = e;
            rsum += e;
        }
        #pragma unroll
        for (int off = 1; off < 16; off <<= 1)
            rsum += __shfl_xor(rsum, off, 32);

        const float inv = DROP_SCALE / rsum;
        #pragma unroll
        for (int st = 0; st < 8; ++st) {
            int scol = st * 16 + lmod;
            unsigned hsh = hash_u32(((unsigned)b << 14) | ((unsigned)t << 7) | (unsigned)scol);
            float val = (hsh < DROP_KEEP_THRESH) ? s[st][g] * inv : 0.0f;
            ps[t * T_DIM + scol] = (_Float16)val;      // own-wave region, in-order DS
        }
    }

    // ------------------- Phase 3: O = P @ v ----------------------------------
    half16 ap[4];
    #pragma unroll
    for (int kk = 0; kk < 4; ++kk)
        ap[kk] = load_a_frag(ps, 16 * wave, kk * 32, T_DIM, lmod, lane_half);

    #pragma unroll
    for (int nt = 0; nt < 4; ++nt) {
        floatx8 acc = {};
        #pragma unroll
        for (int kk = 0; kk < 4; ++kk) {
            // B[K=s][N=h] = vT[h][s]: lane N = nt*16+lmod, K = kk*32+lane_half*16+j
            half16 bb = *(const half16*)(vts + (nt * 16 + lmod) * T_DIM
                                             + kk * 32 + lane_half * 16);
            acc = WMMA_F16(ap[kk], bb, acc);
        }
        #pragma unroll
        for (int g = 0; g < 8; ++g) {
            int t = 16 * wave + g + 8 * lane_half;
            int h = nt * 16 + lmod;
            out[((size_t)b * T_DIM + t) * H_DIM + h] = acc[g];
        }
    }
}

// ---------------------------------------------------------------------------
extern "C" void kernel_launch(void* const* d_in, const int* in_sizes, int n_in,
                              void* d_out, int out_size, void* d_ws, size_t ws_size,
                              hipStream_t stream) {
    const float* x  = (const float*)d_in[0];
    const float* Wq = (const float*)d_in[1];
    const float* Wk = (const float*)d_in[2];
    const float* Wv = (const float*)d_in[3];
    float* out = (float*)d_out;

    _Float16* wT = (_Float16*)d_ws;   // [3][64][256] f16 = 96 KB

    prep_weights_kernel<<<(3 * H_DIM * E_DIM) / 256, 256, 0, stream>>>(Wq, Wk, Wv, wT);

    // LDS: qs(16K) + ks(16K) + vts(16K) + ps(32K) = 80 KB (gfx1250 WGP: 320 KB)
    const size_t smem_bytes = (size_t)(3 * T_DIM * H_DIM + T_DIM * T_DIM) * sizeof(_Float16);
    attention_kernel<<<B_DIM, 256, smem_bytes, stream>>>(x, wT, out);
}
